// MolGNN2_62758062129300
// MI455X (gfx1250) — matrix operand order repaired and checked by reading, not compiled, not run on hardware
//
#include <hip/hip_runtime.h>
#include <math.h>

#define NNODES 200000
#define NEDGES 400000
#define NG     8192
#define F_IN   64
#define OC     180
#define OCP    192
#define FC     540
#define FCP    544
#define H3     1620
#define H3P    1632
#define NL     6

typedef __attribute__((ext_vector_type(16))) __bf16 v16bf;
typedef __attribute__((ext_vector_type(8)))  __bf16 v8bf;
typedef __attribute__((ext_vector_type(8)))  float  v8f;

__device__ __forceinline__ unsigned short bfbits(float f) {
  unsigned u = __float_as_uint(f);
  u += 0x7FFFu + ((u >> 16) & 1u);          // round-to-nearest-even
  return (unsigned short)(u >> 16);
}
__device__ __forceinline__ __bf16 f2bf(float f) {
  unsigned short s = bfbits(f);
  return __builtin_bit_cast(__bf16, s);
}
__device__ __forceinline__ float sigmoidf(float x) { return 1.f / (1.f + expf(-x)); }

// A-fragment (16x32 bf16) per ISA layout: lanes 0-15 -> M=lane, K in
// [k0+0..7] (elem 0-7) and [k0+16..23] (elem 8-15); lanes 16-31 shifted by 8.
__device__ __forceinline__ v16bf load_a_frag(const __bf16* tile, int stride, int k0, int lane) {
  int m  = lane & 15;
  int kh = (lane >> 4) * 8;
  v8bf lo = *(const v8bf*)(tile + m * stride + k0 + kh);
  v8bf hi = *(const v8bf*)(tile + m * stride + k0 + 16 + kh);
  return __builtin_shufflevector(lo, hi, 0,1,2,3,4,5,6,7,8,9,10,11,12,13,14,15);
}

__device__ __forceinline__ v8f wmma_bf16(v16bf a, v16bf b, v8f c) {
  return __builtin_amdgcn_wmma_f32_16x16x32_bf16(false, a, false, b, (short)0, c, false, false);
}

// ---------------------------------------------------------------- utilities
__global__ void zero_f32(float* p, long long n) {
  long long i = (long long)blockIdx.x * blockDim.x + threadIdx.x;
  if (i < n) p[i] = 0.f;
}

__global__ void pad_bias(const float* __restrict__ b, float* __restrict__ out,
                         int nreal, int npad) {
  int i = blockIdx.x * blockDim.x + threadIdx.x;
  if (i < npad) out[i] = (i < nreal) ? b[i] : 0.f;
}

// Pack weight matrix into WMMA B-fragment layout.
// Tile T = nt*kchunks + kc holds 32 lanes x 16 bf16 (lane = N%16 | (Khalf<<4)).
// transpose==0: src is [Kreal rows, ldsrc cols], B[k][n] = src[k*ldsrc + n]
// transpose==1: src is [Nreal rows, Kreal cols], B[k][n] = src[n*Kreal + k]
__global__ void pack_b(const float* __restrict__ W, unsigned short* __restrict__ out,
                       int Kreal, int Nreal, int ldsrc, int kchunks, int ntiles,
                       int transpose) {
  int total = kchunks * ntiles * 512;
  int idx = blockIdx.x * blockDim.x + threadIdx.x;
  if (idx >= total) return;
  int T = idx >> 9;
  int r = idx & 511;
  int L = r >> 4;
  int j = r & 15;
  int nt = T / kchunks;
  int kc = T - nt * kchunks;
  int n = nt * 16 + (L & 15);
  int k = kc * 32 + (L >> 4) * 8 + (j & 7) + (j >> 3) * 16;
  float v = 0.f;
  if (n < Nreal && k < Kreal)
    v = transpose ? W[(size_t)n * Kreal + k] : W[(size_t)k * ldsrc + n];
  out[idx] = bfbits(v);
}

// h[i][c] = (c<64) ? x[i][c] : 0   (pad F_IN=64 -> OCP=192)
__global__ void cast_pad(const float* __restrict__ x, float* __restrict__ h) {
  long long idx = (long long)blockIdx.x * blockDim.x + threadIdx.x;
  if (idx >= (long long)NNODES * OCP) return;
  int i = (int)(idx / OCP), c = (int)(idx % OCP);
  h[idx] = (c < F_IN) ? x[(long long)i * F_IN + c] : 0.f;
}

// ------------------------------------------------------ generic WMMA GEMM
// C[M x ldc] = A[M x lda] (fp32, K zero-padded to Kpad) * Bpacked (+bias, relu)
// block = 256 threads (8 waves), block computes 16 rows x all N tiles.
__global__ void wmma_gemm(const float* __restrict__ A, int lda, int Kpad,
                          const unsigned short* __restrict__ Bp,
                          const float* __restrict__ bias,
                          float* __restrict__ C, int ldc,
                          int kchunks, int ntiles, int relu) {
  extern __shared__ char smem[];
  __bf16* sA = (__bf16*)smem;                       // 16 * Kpad
  int tid = threadIdx.x;
  long long rowBase = (long long)blockIdx.x * 16;
  for (int idx = tid; idx < 16 * Kpad; idx += 256) {
    int m = idx / Kpad, k = idx - m * Kpad;
    sA[idx] = f2bf(A[(rowBase + m) * lda + k]);
  }
  __syncthreads();
  int wave = tid >> 5, lane = tid & 31;
  const v16bf* B16 = (const v16bf*)Bp;
  for (int nt = wave; nt < ntiles; nt += 8) {
    v8f acc = {0.f, 0.f, 0.f, 0.f, 0.f, 0.f, 0.f, 0.f};
    for (int kc = 0; kc < kchunks; kc++) {
      v16bf fa = load_a_frag(sA, Kpad, kc * 32, lane);
      v16bf fb = B16[(size_t)(nt * kchunks + kc) * 32 + lane];
      acc = wmma_bf16(fa, fb, acc);
    }
    int n = nt * 16 + (lane & 15);
    float bv = bias ? bias[n] : 0.f;
    long long m0 = rowBase + ((lane >> 4) * 8);
    #pragma unroll
    for (int i = 0; i < 8; i++) {
      float v = acc[i] + bv;
      if (relu) v = fmaxf(v, 0.f);
      C[(m0 + i) * ldc + n] = v;
    }
  }
}

// ------------------------------------------------------ edge scatter-add
__global__ void scatter_edges(const int* __restrict__ ei,
                              const float* __restrict__ m,
                              float* __restrict__ agg) {
  int wid  = (int)((blockIdx.x * blockDim.x + threadIdx.x) >> 5);
  int lane = threadIdx.x & 31;
  if (wid >= NEDGES) return;
  int src = ei[wid];
  int dst = ei[NEDGES + wid];
  const float* ms = m   + (size_t)src * OCP;
  float*       ad = agg + (size_t)dst * OCP;
  for (int c = lane; c < OC; c += 32) atomicAdd(ad + c, ms[c]);
}

// ------------------------------------------------------ fused GRU cell
// 16-row tile: gi = agg@W_ih^T+b, gh = h@W_hh^T+b via WMMA into LDS, then
// elementwise gate math, h updated in place.
__global__ void gru_fused(float* __restrict__ h, const float* __restrict__ agg,
                          const unsigned short* __restrict__ Wih,
                          const unsigned short* __restrict__ Whh,
                          const float* __restrict__ bih,
                          const float* __restrict__ bhh) {
  extern __shared__ char smem[];
  __bf16* sAgg = (__bf16*)smem;                     // 16*OCP bf16
  __bf16* sH   = sAgg + 16 * OCP;                   // 16*OCP bf16
  float*  sGi  = (float*)(sH + 16 * OCP);           // 16*FCP f32
  float*  sGh  = sGi + 16 * FCP;                    // 16*FCP f32
  int tid = threadIdx.x;
  long long rowBase = (long long)blockIdx.x * 16;
  for (int idx = tid; idx < 16 * OCP; idx += 256) {
    int m = idx / OCP, k = idx - m * OCP;
    long long o = (rowBase + m) * OCP + k;
    sAgg[idx] = f2bf(agg[o]);
    sH[idx]   = f2bf(h[o]);
  }
  __syncthreads();
  int wave = tid >> 5, lane = tid & 31;
  const v16bf* Bi = (const v16bf*)Wih;
  const v16bf* Bh = (const v16bf*)Whh;
  const int kchunks = OCP / 32;                     // 6
  const int ntiles  = FCP / 16;                     // 34
  for (int nt = wave; nt < ntiles; nt += 8) {
    v8f ai = {0.f, 0.f, 0.f, 0.f, 0.f, 0.f, 0.f, 0.f};
    v8f ah = {0.f, 0.f, 0.f, 0.f, 0.f, 0.f, 0.f, 0.f};
    for (int kc = 0; kc < kchunks; kc++) {
      v16bf fa  = load_a_frag(sAgg, OCP, kc * 32, lane);
      v16bf fh  = load_a_frag(sH,   OCP, kc * 32, lane);
      v16bf fbi = Bi[(size_t)(nt * kchunks + kc) * 32 + lane];
      v16bf fbh = Bh[(size_t)(nt * kchunks + kc) * 32 + lane];
      ai = wmma_bf16(fa, fbi, ai);
      ah = wmma_bf16(fh, fbh, ah);
    }
    int n  = nt * 16 + (lane & 15);
    int m0 = (lane >> 4) * 8;
    float vbi = bih[n], vbh = bhh[n];
    #pragma unroll
    for (int i = 0; i < 8; i++) {
      sGi[(m0 + i) * FCP + n] = ai[i] + vbi;
      sGh[(m0 + i) * FCP + n] = ah[i] + vbh;
    }
  }
  __syncthreads();
  for (int idx = tid; idx < 16 * OC; idx += 256) {
    int m = idx / OC, c = idx - m * OC;
    float r  = sigmoidf(sGi[m * FCP + c]          + sGh[m * FCP + c]);
    float z  = sigmoidf(sGi[m * FCP + c + OC]     + sGh[m * FCP + c + OC]);
    float nn = tanhf   (sGi[m * FCP + c + 2 * OC] + r * sGh[m * FCP + c + 2 * OC]);
    long long o = (rowBase + m) * OCP + c;
    float ho = h[o];
    h[o] = (1.f - z) * nn + z * ho;
  }
}

// ------------------------------------------------------ pooling
__global__ void pool_scatter(const float* __restrict__ h, const int* __restrict__ batch,
                             float* __restrict__ sums, float* __restrict__ cnt) {
  int wid  = (int)((blockIdx.x * blockDim.x + threadIdx.x) >> 5);
  int lane = threadIdx.x & 31;
  if (wid >= NNODES) return;
  int b = batch[wid];
  const float* hr = h + (size_t)wid * OCP;
  float* sr = sums + (size_t)b * OC;
  for (int c = lane; c < OC; c += 32) atomicAdd(sr + c, fmaxf(hr[c], 0.f));
  if (lane == 0) atomicAdd(cnt + b, 1.f);
}

__global__ void pool_div(const float* __restrict__ sums, const float* __restrict__ cnt,
                         float* __restrict__ xcat, int colOff) {
  int idx = blockIdx.x * blockDim.x + threadIdx.x;
  if (idx >= NG * OC) return;
  int g = idx / OC, c = idx - g * OC;
  xcat[(size_t)g * FCP + colOff + c] = sums[idx] / fmaxf(cnt[g], 1.f);
}

// ------------------------------------------------------ batchnorm + head
__global__ void bn_stats(const float* __restrict__ x, float* __restrict__ mu,
                         float* __restrict__ rs) {
  __shared__ float s1[256], s2[256];
  int c = blockIdx.x, t = threadIdx.x;
  float a = 0.f, b = 0.f;
  for (int g = t; g < NG; g += 256) {
    float v = x[(size_t)g * FCP + c];
    a += v; b += v * v;
  }
  s1[t] = a; s2[t] = b;
  __syncthreads();
  for (int s = 128; s > 0; s >>= 1) {
    if (t < s) { s1[t] += s1[t + s]; s2[t] += s2[t + s]; }
    __syncthreads();
  }
  if (t == 0) {
    float m = s1[0] / (float)NG;
    float v = s2[0] / (float)NG - m * m;
    mu[c] = m;
    rs[c] = rsqrtf(v + 1e-5f);
  }
}

__global__ void bn_apply(const float* __restrict__ x, const float* __restrict__ mu,
                         const float* __restrict__ rs, const float* __restrict__ gamma,
                         const float* __restrict__ beta, float* __restrict__ out) {
  long long idx = (long long)blockIdx.x * blockDim.x + threadIdx.x;
  if (idx >= (long long)NG * FCP) return;
  int c = (int)(idx % FCP);
  out[idx] = (c < FC) ? (x[idx] - mu[c]) * rs[c] * gamma[c] + beta[c] : 0.f;
}

__global__ void fc3_kernel(const float* __restrict__ a, const float* __restrict__ w,
                           const float* __restrict__ b, float* __restrict__ out) {
  int idx = blockIdx.x * blockDim.x + threadIdx.x;
  if (idx >= NG * 3) return;
  int g = idx / 3, o = idx - g * 3;
  const float* ar = a + (size_t)g * FCP;
  const float* wr = w + (size_t)o * FC;
  float s = b[o];
  for (int k = 0; k < FC; k++) s += ar[k] * wr[k];
  out[idx] = s;
}

// ---------------------------------------------------------------- launch
extern "C" void kernel_launch(void* const* d_in, const int* in_sizes, int n_in,
                              void* d_out, int out_size, void* d_ws, size_t ws_size,
                              hipStream_t stream) {
  (void)in_sizes; (void)n_in; (void)out_size; (void)ws_size;
  char* p = (char*)d_ws;
  auto carve = [&](size_t bytes) -> char* {
    char* r = p; p += (bytes + 255) & ~(size_t)255; return r;
  };
  float* hBuf   = (float*)carve((size_t)NNODES * OCP * 4);
  float* mBuf   = (float*)carve((size_t)NNODES * OCP * 4);
  float* aggBuf = (float*)carve((size_t)NNODES * OCP * 4);
  unsigned short* WlP  = (unsigned short*)carve((size_t)NL * 6 * 12 * 512 * 2);
  unsigned short* WihP = (unsigned short*)carve((size_t)6 * 34 * 512 * 2);
  unsigned short* WhhP = (unsigned short*)carve((size_t)6 * 34 * 512 * 2);
  float* bihP = (float*)carve(FCP * 4);
  float* bhhP = (float*)carve(FCP * 4);
  float* sums = (float*)carve((size_t)NG * OC * 4);
  float* cnt  = (float*)carve((size_t)NG * 4);
  float* xcat = (float*)carve((size_t)NG * FCP * 4);
  float* muB  = (float*)carve(FCP * 4);
  float* rsB  = (float*)carve(FCP * 4);
  float* xbn  = (float*)carve((size_t)NG * FCP * 4);
  float* y1   = (float*)carve((size_t)NG * H3P * 4);
  float* y2   = (float*)carve((size_t)NG * H3P * 4);
  float* y25  = (float*)carve((size_t)NG * FCP * 4);
  unsigned short* f1P  = (unsigned short*)carve((size_t)17 * 102 * 512 * 2);
  unsigned short* f2P  = (unsigned short*)carve((size_t)51 * 102 * 512 * 2);
  unsigned short* f25P = (unsigned short*)carve((size_t)51 * 34 * 512 * 2);
  float* f1b  = (float*)carve(H3P * 4);
  float* f2b  = (float*)carve(H3P * 4);
  float* f25b = (float*)carve(FCP * 4);

  dim3 B(256);
  auto blocks = [](long long n) { return dim3((unsigned)((n + 255) / 256)); };

  zero_f32<<<blocks((long long)NG * FCP), B, 0, stream>>>(xcat, (long long)NG * FCP);

  // head weight packing (fcN_w is [out, in], used transposed)
  pack_b<<<blocks(17 * 102 * 512), B, 0, stream>>>((const float*)d_in[26], f1P, FC, H3, 0, 17, 102, 1);
  pack_b<<<blocks(51 * 102 * 512), B, 0, stream>>>((const float*)d_in[28], f2P, H3, H3, 0, 51, 102, 1);
  pack_b<<<blocks(51 * 34 * 512),  B, 0, stream>>>((const float*)d_in[30], f25P, H3, FC, 0, 51, 34, 1);
  pad_bias<<<blocks(H3P), B, 0, stream>>>((const float*)d_in[27], f1b, H3, H3P);
  pad_bias<<<blocks(H3P), B, 0, stream>>>((const float*)d_in[29], f2b, H3, H3P);
  pad_bias<<<blocks(FCP), B, 0, stream>>>((const float*)d_in[31], f25b, FC, FCP);

  for (int comp = 0; comp < 3; comp++) {
    const float* x   = (const float*)d_in[comp * 8 + 0];
    const int*   ei  = (const int*)  d_in[comp * 8 + 1];
    const int*   bat = (const int*)  d_in[comp * 8 + 2];
    const float* Wl  = (const float*)d_in[comp * 8 + 3];
    const float* wih = (const float*)d_in[comp * 8 + 4];
    const float* whh = (const float*)d_in[comp * 8 + 5];
    const float* bih = (const float*)d_in[comp * 8 + 6];
    const float* bhh = (const float*)d_in[comp * 8 + 7];

    for (int l = 0; l < NL; l++)
      pack_b<<<blocks(6 * 12 * 512), B, 0, stream>>>(
          Wl + (size_t)l * OC * OC, WlP + (size_t)l * 6 * 12 * 512, OC, OC, OC, 6, 12, 0);
    pack_b<<<blocks(6 * 34 * 512), B, 0, stream>>>(wih, WihP, OC, FC, 0, 6, 34, 1);
    pack_b<<<blocks(6 * 34 * 512), B, 0, stream>>>(whh, WhhP, OC, FC, 0, 6, 34, 1);
    pad_bias<<<blocks(FCP), B, 0, stream>>>(bih, bihP, FC, FCP);
    pad_bias<<<blocks(FCP), B, 0, stream>>>(bhh, bhhP, FC, FCP);

    cast_pad<<<blocks((long long)NNODES * OCP), B, 0, stream>>>(x, hBuf);

    for (int l = 0; l < NL; l++) {
      // m = h @ W_l
      wmma_gemm<<<dim3(NNODES / 16), B, 16 * OCP * 2, stream>>>(
          hBuf, OCP, OCP, WlP + (size_t)l * 6 * 12 * 512, nullptr, mBuf, OCP, 6, 12, 0);
      // agg = scatter_add over edges
      zero_f32<<<blocks((long long)NNODES * OCP), B, 0, stream>>>(aggBuf, (long long)NNODES * OCP);
      scatter_edges<<<dim3(NEDGES / 8), B, 0, stream>>>(ei, mBuf, aggBuf);
      // h = GRU(agg, h)
      gru_fused<<<dim3(NNODES / 16), B, 2 * 16 * OCP * 2 + 2 * 16 * FCP * 4, stream>>>(
          hBuf, aggBuf, WihP, WhhP, bihP, bhhP);
    }
    zero_f32<<<blocks((long long)NG * OC), B, 0, stream>>>(sums, (long long)NG * OC);
    zero_f32<<<blocks(NG), B, 0, stream>>>(cnt, NG);
    pool_scatter<<<dim3(NNODES / 8), B, 0, stream>>>(hBuf, bat, sums, cnt);
    pool_div<<<blocks((long long)NG * OC), B, 0, stream>>>(sums, cnt, xcat, comp * OC);
  }

  bn_stats<<<dim3(FC), B, 0, stream>>>(xcat, muB, rsB);
  bn_apply<<<blocks((long long)NG * FCP), B, 0, stream>>>(
      xcat, muB, rsB, (const float*)d_in[24], (const float*)d_in[25], xbn);
  wmma_gemm<<<dim3(NG / 16), B, 16 * FCP * 2, stream>>>(xbn, FCP, FCP, f1P, f1b, y1, H3P, 17, 102, 1);
  wmma_gemm<<<dim3(NG / 16), B, 16 * H3P * 2, stream>>>(y1, H3P, H3P, f2P, f2b, y2, H3P, 51, 102, 1);
  wmma_gemm<<<dim3(NG / 16), B, 16 * H3P * 2, stream>>>(y2, H3P, H3P, f25P, f25b, y25, FCP, 51, 34, 1);
  fc3_kernel<<<blocks((long long)NG * 3), B, 0, stream>>>(
      y25, (const float*)d_in[32], (const float*)d_in[33], (float*)d_out);
}
